// TransformerEncoderLayer_87033217286165
// MI455X (gfx1250) — matrix operand order repaired and checked by reading, not compiled, not run on hardware
//
#include <hip/hip_runtime.h>

typedef float  v8f  __attribute__((ext_vector_type(8)));
typedef __bf16 bf16_t;
typedef __bf16 v16bf __attribute__((ext_vector_type(16)));
typedef __bf16 v8bf  __attribute__((ext_vector_type(8)));

#define B_   8
#define S_   2048
#define D_   512
#define DK_  32
#define NINF_ (-1e24f)
#define EPS_  1e-5f

__device__ __forceinline__ v8f wmma_bf16(v16bf a, v16bf b, v8f c) {
    // (neg_a, A, neg_b, B, c_mod, C, reuse_a, reuse_b)
    return __builtin_amdgcn_wmma_f32_16x16x32_bf16(false, a, false, b, (short)0, c,
                                                   false, false);
}

union V16U { v16bf v; v8bf h[2]; };
// Assemble a 16-element A/B fragment from two aligned 16-byte chunks.
__device__ __forceinline__ v16bf load_frag(const bf16_t* p0, const bf16_t* p1) {
    V16U u;
    u.h[0] = *(const v8bf*)p0;
    u.h[1] = *(const v8bf*)p1;
    return u.v;
}

// ---------------------------------------------------------------------------
// fp32 -> bf16 bulk convert (n multiple of 4)
// ---------------------------------------------------------------------------
typedef __bf16 v4bf __attribute__((ext_vector_type(4)));
__global__ void cvt_f32_bf16(const float* __restrict__ in, bf16_t* __restrict__ out,
                             int n4) {
    int i = blockIdx.x * blockDim.x + threadIdx.x;
    if (i >= n4) return;
    const float4 f = ((const float4*)in)[i];
    v4bf o;
    o.x = (bf16_t)f.x; o.y = (bf16_t)f.y; o.z = (bf16_t)f.z; o.w = (bf16_t)f.w;
    ((v4bf*)out)[i] = o;
}

// ---------------------------------------------------------------------------
// W[K,N] fp32 -> Wt[N,K] bf16 (weights are tiny; stays in L2)
// ---------------------------------------------------------------------------
__global__ void transpose_w_bf16(const float* __restrict__ W, bf16_t* __restrict__ Wt,
                                 int K, int N) {
    int i = blockIdx.x * blockDim.x + threadIdx.x;
    if (i >= K * N) return;
    int n = i / K, k = i - n * K;
    Wt[i] = (bf16_t)W[(size_t)k * N + n];
}

// ---------------------------------------------------------------------------
// bf16 GEMM: C[M,N] = A[M,K] @ W[K,N], A bf16 row-major, Wt = W^T bf16 [N,K].
// One wave per 16x16 C tile, 8 waves per block, fp32 accumulation.
// mode 0: Cf fp32 (+bias)      mode 1: Cb bf16
// mode 2: Cb bf16, key-pad rows forced to -1e24
// mode 3: Cb bf16 stored per-batch transposed: vt[(b*D + col)*S + s]
// ---------------------------------------------------------------------------
__global__ void gemm_bf16_wmma(const bf16_t* __restrict__ A,
                               const bf16_t* __restrict__ Wt,
                               float* __restrict__ Cf, bf16_t* __restrict__ Cb,
                               int M, int N, int K, int numTiles, int mode,
                               const float* __restrict__ bias,
                               const unsigned char* __restrict__ rowmask) {
    const int wave = threadIdx.x >> 5;
    const int tile = blockIdx.x * 8 + wave;
    if (tile >= numTiles) return;                    // wave-uniform exit
    const int ntilesN = N >> 4;
    const int tm = tile / ntilesN;
    const int tn = tile - tm * ntilesN;
    const int row0 = tm << 4, col0 = tn << 4;
    const int lane = threadIdx.x & 31;
    const int half = lane >> 4;
    const int l15  = lane & 15;

    v8f acc = {};
    // A frag (16-bit A layout): lane m holds K = [k0+8h .. +7] and [k0+16+8h .. +7]
    const bf16_t* Arow = A  + (size_t)(row0 + l15) * K + 8 * half;
    // B frag (16-bit B layout): lane n holds K = [k0+8h .. +7] and [k0+16+8h .. +7] of column n
    const bf16_t* Brow = Wt + (size_t)(col0 + l15) * K + 8 * half;

    for (int k0 = 0; k0 < K; k0 += 32) {
        v16bf a = load_frag(Arow + k0, Arow + k0 + 16);
        v16bf b = load_frag(Brow + k0, Brow + k0 + 16);
        acc = wmma_bf16(a, b, acc);
    }

    const float bc = (mode == 0 && bias) ? bias[col0 + l15] : 0.0f;
#pragma unroll
    for (int r = 0; r < 8; ++r) {
        const int m = row0 + r + 8 * half;           // C layout: VGPR r -> row r / 8+r
        const int col = col0 + l15;
        float v = acc[r] + bc;
        if (mode == 0) {
            Cf[(size_t)m * N + col] = v;
        } else if (mode == 1) {
            Cb[(size_t)m * N + col] = (bf16_t)v;
        } else if (mode == 2) {
            if (rowmask[m]) v = NINF_;
            Cb[(size_t)m * N + col] = (bf16_t)v;
        } else {                                     // mode 3: v^T per batch
            const int b = m >> 11;                   // m / S_
            const int s = m & (S_ - 1);
            Cb[((size_t)b * D_ + col) * S_ + s] = (bf16_t)v;
        }
    }
}

// ---------------------------------------------------------------------------
// Flash attention, bf16 WMMA (fp32 accum). One block per (batch, 16-query
// tile); 8 waves, wave w owns output columns [64w, 64w+64). Keys processed
// 32 at a time: 2 score WMMAs (16x16x32, K = DK = 32) + streaming softmax +
// 4 P@V WMMAs. P is converted bf16 and transposed C->A layout through a
// per-wave LDS bounce (same-wave DS ops are in-order; no barrier needed).
// v is pre-transposed: vt[(b*D + d)*S + s].
// ---------------------------------------------------------------------------
__global__ void attn_flash_wmma(const bf16_t* __restrict__ q,
                                const bf16_t* __restrict__ k,
                                const bf16_t* __restrict__ vt,
                                float* __restrict__ o) {
    __shared__ bf16_t pbuf[8 * 16 * 40];             // stride 40: conflict-free b128
    const int lane = threadIdx.x & 31;
    const int wave = threadIdx.x >> 5;
    const int half = lane >> 4;
    const int l15  = lane & 15;

    const int b  = blockIdx.x >> 7;                  // S_/16 = 128 query tiles
    const int qt = blockIdx.x & 127;
    const int qrow0 = qt << 4;
    const int colbase = wave * 64;

    const bf16_t* qb = q  + (size_t)b * S_ * DK_;
    const bf16_t* kb = k  + (size_t)b * S_ * DK_;
    const bf16_t* vb = vt + (size_t)b * D_ * S_;
    float*        ob = o  + (size_t)b * S_ * D_;

    // Resident Q A-fragment (16x32, K = DK): one v16bf
    const bf16_t* qrow = qb + (size_t)(qrow0 + l15) * DK_;
    const v16bf qa = load_frag(qrow + 8 * half, qrow + 16 + 8 * half);

    v8f accO[4];
#pragma unroll
    for (int t = 0; t < 4; ++t) accO[t] = (v8f){};
    float mr[8], lr[8];
#pragma unroll
    for (int r = 0; r < 8; ++r) { mr[r] = -INFINITY; lr[r] = 0.0f; }

    const float scale = 0.17677669529663687f;        // 1/sqrt(DK)
    bf16_t* pl = &pbuf[wave * 16 * 40];

    for (int kt = 0; kt <= qt; kt += 2) {            // 32-key chunks, causal
        const int kb0 = kt << 4;
        const int kb1 = kb0 + 16;
        const bool valid1 = (kt + 1) <= qt;          // block-uniform

        // ---- scores: one 16x16x32 WMMA per 16-key subtile ----
        v8f s0 = {}, s1 = {};
        {
            const bf16_t* kr = kb + (size_t)(kb0 + l15) * DK_ + 8 * half;
            s0 = wmma_bf16(qa, load_frag(kr, kr + 16), s0);
        }
        if (valid1) {
            const bf16_t* kr = kb + (size_t)(kb1 + l15) * DK_ + 8 * half;
            s1 = wmma_bf16(qa, load_frag(kr, kr + 16), s1);
        }

        // ---- online softmax over the 32-key chunk ----
        float alpha[8];
#pragma unroll
        for (int r = 0; r < 8; ++r) {
            const int qm = r + 8 * half;
            float sv0 = s0[r] * scale;
            if (kt == qt && l15 > qm) sv0 = NINF_;   // diagonal causal mask
            float sv1 = NINF_;
            if (valid1) {
                sv1 = s1[r] * scale;
                if (kt + 1 == qt && l15 > qm) sv1 = NINF_;
            }
            float mx = fmaxf(sv0, sv1);              // row-max over 16 lanes x 2
            mx = fmaxf(mx, __shfl_xor(mx, 1));
            mx = fmaxf(mx, __shfl_xor(mx, 2));
            mx = fmaxf(mx, __shfl_xor(mx, 4));
            mx = fmaxf(mx, __shfl_xor(mx, 8));
            const float mnew = fmaxf(mr[r], mx);
            const float a  = __expf(mr[r] - mnew);
            const float p0 = __expf(sv0 - mnew);
            const float p1 = __expf(sv1 - mnew);
            float ps = p0 + p1;
            ps += __shfl_xor(ps, 1);
            ps += __shfl_xor(ps, 2);
            ps += __shfl_xor(ps, 4);
            ps += __shfl_xor(ps, 8);
            lr[r] = lr[r] * a + ps;
            mr[r] = mnew;
            alpha[r] = a;
            // P tile -> LDS as bf16, row-major 16x32 (stride 40)
            pl[qm * 40 + l15]      = (bf16_t)p0;
            pl[qm * 40 + 16 + l15] = (bf16_t)p1;
        }
#pragma unroll
        for (int t = 0; t < 4; ++t)
#pragma unroll
            for (int r = 0; r < 8; ++r) accO[t][r] *= alpha[r];

        // ---- reload P in A layout (two aligned b128 per lane) ----
        const bf16_t* prow = pl + l15 * 40;
        const v16bf pa = load_frag(prow + 8 * half, prow + 16 + 8 * half);

        // ---- O += P @ V : one 16x16x32 WMMA per 16-col tile ----
#pragma unroll
        for (int t = 0; t < 4; ++t) {
            const bf16_t* vr = vb + (size_t)(colbase + t * 16 + l15) * S_
                                  + kb0 + 8 * half;
            accO[t] = wmma_bf16(pa, load_frag(vr, vr + 16), accO[t]);
        }
    }

    float invl[8];
#pragma unroll
    for (int r = 0; r < 8; ++r) invl[r] = 1.0f / lr[r];
#pragma unroll
    for (int t = 0; t < 4; ++t)
#pragma unroll
        for (int r = 0; r < 8; ++r) {
            const int row = qrow0 + r + 8 * half;
            ob[(size_t)row * D_ + colbase + t * 16 + l15] = accO[t][r] * invl[r];
        }
}

// ---------------------------------------------------------------------------
// out = LayerNorm(a + b) * g + beta (fp32); optionally also emit bf16 copy.
// One wave per 512-wide row; 8 rows/block.
// ---------------------------------------------------------------------------
__global__ void add_layernorm(const float* __restrict__ a,
                              const float* __restrict__ bres,
                              const float* __restrict__ g,
                              const float* __restrict__ beta,
                              float* __restrict__ out,
                              bf16_t* __restrict__ out_bf,
                              int nrows) {
    const int wave = threadIdx.x >> 5;
    const int lane = threadIdx.x & 31;
    const int row = blockIdx.x * 8 + wave;
    if (row >= nrows) return;
    const float* ar = a    + (size_t)row * D_;
    const float* br = bres + (size_t)row * D_;

    float vals[16];
    float sum = 0.0f;
#pragma unroll
    for (int i = 0; i < 16; ++i) {
        float t = ar[lane + 32 * i] + br[lane + 32 * i];
        vals[i] = t;
        sum += t;
    }
#pragma unroll
    for (int m = 1; m <= 16; m <<= 1) sum += __shfl_xor(sum, m);
    const float mu = sum * (1.0f / (float)D_);

    float vs = 0.0f;
#pragma unroll
    for (int i = 0; i < 16; ++i) { float d = vals[i] - mu; vs += d * d; }
#pragma unroll
    for (int m = 1; m <= 16; m <<= 1) vs += __shfl_xor(vs, m);
    const float rstd = rsqrtf(vs * (1.0f / (float)D_) + EPS_);

    float*  orow = out + (size_t)row * D_;
    bf16_t* brow = out_bf ? out_bf + (size_t)row * D_ : nullptr;
#pragma unroll
    for (int i = 0; i < 16; ++i) {
        const int c = lane + 32 * i;
        const float r = (vals[i] - mu) * rstd * g[c] + beta[c];
        orow[c] = r;
        if (brow) brow[c] = (bf16_t)r;
    }
}

// ---------------------------------------------------------------------------
extern "C" void kernel_launch(void* const* d_in, const int* in_sizes, int n_in,
                              void* d_out, int out_size, void* d_ws, size_t ws_size,
                              hipStream_t stream) {
    const float* x   = (const float*)d_in[0];
    // d_in[1] = attn_mask (causal triu) -- structure known, applied analytically
    const unsigned char* kpm = (const unsigned char*)d_in[2];  // (B,S,1) bool
    const float* Wq  = (const float*)d_in[3];
    const float* Wk  = (const float*)d_in[4];
    const float* Wv  = (const float*)d_in[5];
    const float* Wff = (const float*)d_in[6];
    const float* bff = (const float*)d_in[7];
    const float* g1  = (const float*)d_in[8];
    const float* b1  = (const float*)d_in[9];
    const float* g2  = (const float*)d_in[10];
    const float* b2  = (const float*)d_in[11];
    float* out = (float*)d_out;

    const size_t MS = (size_t)B_ * S_;               // 16384 rows
    const size_t XE = MS * D_;                       // 8388608 activation elems

    // -------- workspace carve-up (bf16 region first, 128B-aligned total) ----
    bf16_t* x_bf  = (bf16_t*)d_ws;                   // XE
    bf16_t* Wq_t  = x_bf  + XE;                      // DK*D
    bf16_t* Wk_t  = Wq_t  + (size_t)DK_ * D_;        // DK*D
    bf16_t* Wv_t  = Wk_t  + (size_t)DK_ * D_;        // D*D
    bf16_t* Wff_t = Wv_t  + (size_t)D_ * D_;         // D*D
    bf16_t* q_bf  = Wff_t + (size_t)D_ * D_;         // MS*DK
    bf16_t* k_bf  = q_bf  + MS * DK_;                // MS*DK
    bf16_t* vT    = k_bf  + MS * DK_;                // B*D*S (+64 pad: causal tail)
    bf16_t* z_bf  = vT    + XE + 64;                 // XE
    float*  attn  = (float*)(z_bf + XE);             // XE fp32 (reused as FF out)
    float*  z     = attn + XE;                       // XE fp32

    const int M = (int)MS;
    const dim3 blk(256);

    // ---- precision staging: x -> bf16; weights -> bf16 transposed [N,K] ----
    cvt_f32_bf16<<<(int)(XE / 4 + 255) / 256, blk, 0, stream>>>(x, x_bf, (int)(XE / 4));
    transpose_w_bf16<<<(DK_ * D_ + 255) / 256, blk, 0, stream>>>(Wq,  Wq_t,  D_, DK_);
    transpose_w_bf16<<<(DK_ * D_ + 255) / 256, blk, 0, stream>>>(Wk,  Wk_t,  D_, DK_);
    transpose_w_bf16<<<(D_  * D_ + 255) / 256, blk, 0, stream>>>(Wv,  Wv_t,  D_, D_);
    transpose_w_bf16<<<(D_  * D_ + 255) / 256, blk, 0, stream>>>(Wff, Wff_t, D_, D_);

    // ---- QKV projections (bf16 WMMA, fp32 accum) ----
    {
        int nt = (M / 16) * (DK_ / 16);
        gemm_bf16_wmma<<<(nt + 7) / 8, blk, 0, stream>>>(x_bf, Wq_t, nullptr, q_bf,
                                                         M, DK_, D_, nt, 1, nullptr, nullptr);
        gemm_bf16_wmma<<<(nt + 7) / 8, blk, 0, stream>>>(x_bf, Wk_t, nullptr, k_bf,
                                                         M, DK_, D_, nt, 2, nullptr, kpm);
    }
    {
        int nt = (M / 16) * (D_ / 16);
        gemm_bf16_wmma<<<(nt + 7) / 8, blk, 0, stream>>>(x_bf, Wv_t, nullptr, vT,
                                                         M, D_, D_, nt, 3, nullptr, nullptr);
    }

    // ---- fused causal attention (scores never touch HBM) ----
    attn_flash_wmma<<<B_ * (S_ / 16), 256, 0, stream>>>(q_bf, k_bf, vT, attn);

    // ---- z = LN(x + attn), also emit z in bf16 for the FF GEMM ----
    add_layernorm<<<(M + 7) / 8, 256, 0, stream>>>(x, attn, g1, b1, z, z_bf, M);

    // ---- ff = z @ Wff + bff (fp32 out, reuse attn buffer) ----
    {
        int nt = (M / 16) * (D_ / 16);
        gemm_bf16_wmma<<<(nt + 7) / 8, blk, 0, stream>>>(z_bf, Wff_t, attn, nullptr,
                                                         M, D_, D_, nt, 0, bff, nullptr);
    }

    // ---- out = LN(z + ff) ----
    add_layernorm<<<(M + 7) / 8, 256, 0, stream>>>(z, attn, g2, b2, out, nullptr, M);
}
